// Transformer_80307298501045
// MI455X (gfx1250) — matrix operand order repaired and checked by reading, not compiled, not run on hardware
//
#include <hip/hip_runtime.h>

// ---------------------------------------------------------------------------
// Types for CDNA5 WMMA (wave32, v_wmma_f32_16x16x32_bf16)
// ---------------------------------------------------------------------------
typedef __bf16 bf16_t;
typedef __attribute__((ext_vector_type(16))) __bf16 v16bf;
typedef __attribute__((ext_vector_type(8)))  float  v8f;
typedef __attribute__((ext_vector_type(4)))  int    v4i;

#define D_MODEL 512
#define N_HEADS 8
#define DK      64
#define BATCH   4
#define S_SRC   2048
#define S_TGT   1024

// CDNA5 async global->LDS copy path (ASYNCcnt), with sync fallback.
#if defined(__has_builtin)
#if __has_builtin(__builtin_amdgcn_global_load_async_to_lds_b128) && \
    __has_builtin(__builtin_amdgcn_s_wait_asynccnt)
#define USE_ASYNC_COPY 1
#endif
#endif
#ifndef USE_ASYNC_COPY
#define USE_ASYNC_COPY 0
#endif

__device__ __forceinline__ void cp16(bf16_t* dst, const bf16_t* src) {
#if USE_ASYNC_COPY
  __builtin_amdgcn_global_load_async_to_lds_b128(
      (__attribute__((address_space(1))) v4i*)src,
      (__attribute__((address_space(3))) v4i*)dst, 0, 0);
#else
  *(uint4*)dst = *(const uint4*)src;
#endif
}
// Wait until at most N async copies of this wave remain in flight. Async loads
// complete in order per wave, so N = copies-just-issued pipelines the previous
// tile's completion against the next tile's fill.
template <int N>
__device__ __forceinline__ void cp_waitN() {
#if USE_ASYNC_COPY
  __builtin_amdgcn_s_wait_asynccnt(N);
#endif
}

// 16-bit operand layouts (ISA 7.12.2) are piecewise K-contiguous per lane:
//  A-frag: e=0..7  -> K = hv*8 + e ; e=8..15 -> K = 16 + hv*8 + (e-8)
//  B-frag: e=0..7  -> K = hv*8 + e ; e=8..15 -> K = 16 + hv*8 + (e-8)
// With K-contiguous LDS storage each frag is two ds_load_b128.
union FragU { v16bf v; uint4 q[2]; };

__device__ __forceinline__ v16bf load_a_frag(const bf16_t* rowK, int hv) {
  FragU f;
  f.q[0] = *(const uint4*)(rowK + hv * 8);
  f.q[1] = *(const uint4*)(rowK + 16 + hv * 8);
  return f.v;
}
__device__ __forceinline__ v16bf load_b_frag(const bf16_t* colK, int hv) {
  FragU f;
  f.q[0] = *(const uint4*)(colK + hv * 8);
  f.q[1] = *(const uint4*)(colK + 16 + hv * 8);
  return f.v;
}
__device__ __forceinline__ v8f wmma_bf16(v16bf a, v16bf b, v8f c) {
  return __builtin_amdgcn_wmma_f32_16x16x32_bf16(false, a, false, b, (short)0, c,
                                                 false, false);
}
__device__ __forceinline__ v8f v8f_zero() {
  return (v8f){0.f, 0.f, 0.f, 0.f, 0.f, 0.f, 0.f, 0.f};
}

// ---------------------------------------------------------------------------
// fp32 -> bf16 conversion kernels
// ---------------------------------------------------------------------------
__global__ void k_cvt_bf16(const float* __restrict__ in, bf16_t* __restrict__ out,
                           long n) {
  long stride = (long)gridDim.x * blockDim.x;
  for (long i = (long)blockIdx.x * blockDim.x + threadIdx.x; i < n; i += stride)
    out[i] = (bf16_t)in[i];
}

// in[R][C] fp32 -> out[C][R] bf16  (weights stored as W^T[N][K], K-contiguous)
__global__ void k_cvt_bf16_T(const float* __restrict__ in, bf16_t* __restrict__ out,
                             int R, int C) {
  long n = (long)R * C;
  long stride = (long)gridDim.x * blockDim.x;
  for (long i = (long)blockIdx.x * blockDim.x + threadIdx.x; i < n; i += stride) {
    long r = i / C, c = i % C;
    out[c * (long)R + r] = (bf16_t)in[i];
  }
}

// ---------------------------------------------------------------------------
// Embedding + sinusoidal positional encoding; writes fp32 + bf16 activations
// ---------------------------------------------------------------------------
__global__ void k_embed(const int* __restrict__ tok, const float* __restrict__ table,
                        float* __restrict__ xf, bf16_t* __restrict__ xb, int L) {
  int row = blockIdx.x;              // b*L + s
  int s   = row % L;
  int t   = tok[row];
  long base = (long)row * D_MODEL;
  for (int d = threadIdx.x; d < D_MODEL; d += blockDim.x) {
    float div = __expf((float)(d & ~1) * (-9.210340371976184f / 512.0f));
    float ang = (float)s * div;
    float pe  = (d & 1) ? __cosf(ang) : __sinf(ang);
    float v   = table[(long)t * D_MODEL + d] * 22.62741699796952f + pe; // *sqrt(512)
    xf[base + d] = v;
    xb[base + d] = (bf16_t)v;
  }
}

// ---------------------------------------------------------------------------
// Fused residual-add + LayerNorm over 512; writes fp32 (optional) + bf16
// ---------------------------------------------------------------------------
__global__ void k_add_ln(const float* __restrict__ x, const float* __restrict__ a,
                         const float* __restrict__ g, const float* __restrict__ bb,
                         float* __restrict__ of, bf16_t* __restrict__ ob) {
  __shared__ float red[256];
  int row = blockIdx.x, tid = threadIdx.x;
  long base = (long)row * D_MODEL;
  float v0 = x[base + tid], v1 = x[base + tid + 256];
  if (a) { v0 += a[base + tid]; v1 += a[base + tid + 256]; }
  red[tid] = v0 + v1;
  __syncthreads();
  for (int s2 = 128; s2 > 0; s2 >>= 1) {
    if (tid < s2) red[tid] += red[tid + s2];
    __syncthreads();
  }
  float mu = red[0] * (1.0f / 512.0f);
  __syncthreads();
  float d0 = v0 - mu, d1 = v1 - mu;
  red[tid] = d0 * d0 + d1 * d1;
  __syncthreads();
  for (int s2 = 128; s2 > 0; s2 >>= 1) {
    if (tid < s2) red[tid] += red[tid + s2];
    __syncthreads();
  }
  float rs = rsqrtf(red[0] * (1.0f / 512.0f) + 1e-5f);
  float o0 = d0 * rs * g[tid] + bb[tid];
  float o1 = d1 * rs * g[tid + 256] + bb[tid + 256];
  if (of) { of[base + tid] = o0; of[base + tid + 256] = o1; }
  ob[base + tid]       = (bf16_t)o0;
  ob[base + tid + 256] = (bf16_t)o1;
}

// ---------------------------------------------------------------------------
// BF16 WMMA GEMM: C[M,N] = A[M,K] @ W + bias; W given TRANSPOSED as WT[N][K].
// Double-buffered async global->LDS staging: tile t+1 fills one LDS buffer
// (s_wait_asynccnt 4 keeps it in flight) while WMMA consumes the other.
// Block: 256 threads = 8 wave32 waves, tile 128x128, K-step 32.
// ---------------------------------------------------------------------------
__global__ __launch_bounds__(256) void k_gemm(
    const bf16_t* __restrict__ A, const bf16_t* __restrict__ WT,
    const float* __restrict__ bias, float* __restrict__ Cf,
    bf16_t* __restrict__ Cb, bf16_t* __restrict__ CbT, int Lseq,
    int M, int N, int K, int relu) {
  __shared__ __align__(16) bf16_t As[2][128 * 32];    // [m][k]
  __shared__ __align__(16) bf16_t Bst[2][128 * 32];   // [n][k]
  int tid  = threadIdx.x;
  int lane = tid & 31, w = tid >> 5;
  int mw = w >> 1, nw = w & 1;                        // 4x2 wave grid
  int tM = blockIdx.y * 128, tN = blockIdx.x * 128;
  int hv = lane >> 4, lm = lane & 15;

  // Hoisted staging addresses: thread covers rows r0 and r0+64, 8 bf16 each.
  int r0 = tid >> 2, seg = tid & 3;
  int dOff = r0 * 32 + seg * 8;
  const bf16_t* aSrc0 = A  + (long)(tM + r0) * K + seg * 8;
  const bf16_t* aSrc1 = aSrc0 + (long)64 * K;
  const bf16_t* bSrc0 = WT + (long)(tN + r0) * K + seg * 8;
  const bf16_t* bSrc1 = bSrc0 + (long)64 * K;

  auto stage = [&](int k0, int buf) {   // 4 async b128 copies per thread
    cp16(&As[buf][dOff],             aSrc0 + k0);
    cp16(&As[buf][dOff + 64 * 32],   aSrc1 + k0);
    cp16(&Bst[buf][dOff],            bSrc0 + k0);
    cp16(&Bst[buf][dOff + 64 * 32],  bSrc1 + k0);
  };

  v8f acc[2][4];
  for (int i = 0; i < 2; i++)
    for (int j = 0; j < 4; j++) acc[i][j] = v8f_zero();

  stage(0, 0);
  int nt = K >> 5;
  for (int t = 0; t < nt; t++) {
    int buf = t & 1;
    if (t + 1 < nt) {
      stage((t + 1) << 5, buf ^ 1);
      cp_waitN<4>();                  // tile t complete; t+1 still in flight
      __builtin_prefetch(aSrc0 + ((t + 2) << 5), 0, 1);   // L2 warm for t+2
      __builtin_prefetch(bSrc0 + ((t + 2) << 5), 0, 1);
    } else {
      cp_waitN<0>();
    }
    __syncthreads();

    const bf16_t* as = As[buf];
    const bf16_t* bs = Bst[buf];
    v16bf af[2], bfv[4];
    for (int m16 = 0; m16 < 2; m16++)
      af[m16] = load_a_frag(&as[(mw * 32 + m16 * 16 + lm) * 32], hv);
    for (int n16 = 0; n16 < 4; n16++)
      bfv[n16] = load_b_frag(&bs[(nw * 64 + n16 * 16 + lm) * 32], hv);
    for (int m16 = 0; m16 < 2; m16++)
      for (int n16 = 0; n16 < 4; n16++)
        acc[m16][n16] = wmma_bf16(af[m16], bfv[n16], acc[m16][n16]);
    __syncthreads();                  // buf free for tile t+2's fill
  }

  // Epilogue: C/D layout — VGPR r: M = r + 8*(lane/16), N = lane%16.
  for (int m16 = 0; m16 < 2; m16++)
    for (int n16 = 0; n16 < 4; n16++)
      for (int r = 0; r < 8; r++) {
        int gr = tM + mw * 32 + m16 * 16 + r + 8 * hv;
        int gc = tN + nw * 64 + n16 * 16 + lm;
        float v = acc[m16][n16][r] + (bias ? bias[gc] : 0.0f);
        if (relu) v = fmaxf(v, 0.0f);
        if (Cf) Cf[(long)gr * N + gc] = v;
        if (Cb) Cb[(long)gr * N + gc] = (bf16_t)v;
        if (CbT) {  // per-head transposed: [b][h][dk][Lseq]
          int bidx = gr / Lseq, kv = gr % Lseq;
          int hh = gc >> 6, dd = gc & 63;
          CbT[(((long)bidx * N_HEADS + hh) * DK + dd) * Lseq + kv] = (bf16_t)v;
        }
      }
}

// ---------------------------------------------------------------------------
// Flash attention (online softmax), dk = 64, bf16 in/out, f32 accum.
// Q,K: [B*L, 512] row-major; V: per-head transposed [B][H][dk][Lk].
// Block: 128 threads = 4 waves, each wave owns 16 q-rows (q tile = 64 rows).
// K/V kv-tiles double-buffered through async LDS staging (s_wait_asynccnt 8).
// S = Q K^T and O += P V both WMMA; all fragment traffic is ds_load_b128.
// ---------------------------------------------------------------------------
__global__ __launch_bounds__(128) void k_flash(
    const bf16_t* __restrict__ Q, const bf16_t* __restrict__ Kk,
    const bf16_t* __restrict__ Vt, bf16_t* __restrict__ O, int Lq, int Lk,
    int causal, const unsigned char* __restrict__ smask, float scale) {
  __shared__ __align__(16) bf16_t Ql[64 * 64];         // [q][dk]
  __shared__ __align__(16) bf16_t Kl[2][64 * 64];      // [kv][dk]
  __shared__ __align__(16) bf16_t Vl[2][64 * 64];      // [dk][kv]
  __shared__ __align__(16) bf16_t Pl[4][16 * 64];      // per-wave [m][kv]

  int b = blockIdx.z, h = blockIdx.y;
  int q0 = blockIdx.x * 64;
  int tid = threadIdx.x, lane = tid & 31, w = tid >> 5;
  int hv = lane >> 4, lm = lane & 15;

  // Hoisted staging addresses: thread covers rows rr+u*16 (u=0..3), 8 bf16 each.
  int rr = tid >> 3, c8 = (tid & 7) * 8;
  const bf16_t* vhead = Vt + ((long)(b * N_HEADS + h)) * DK * Lk;
  const bf16_t* qBase = Q  + ((long)(b * Lq + q0) + rr) * D_MODEL + h * DK + c8;
  const bf16_t* kBase = Kk + ((long)b * Lk + rr) * D_MODEL + h * DK + c8;
  const bf16_t* vBase = vhead + (long)rr * Lk + c8;

  auto stage_kv = [&](int kv0, int buf) {   // 8 async b128 copies per thread
#pragma unroll
    for (int u = 0; u < 4; u++) {
      cp16(&Kl[buf][(rr + u * 16) * 64 + c8],
           kBase + ((long)kv0 + u * 16) * D_MODEL);
      cp16(&Vl[buf][(rr + u * 16) * 64 + c8], vBase + kv0 + (long)u * 16 * Lk);
    }
  };

#pragma unroll
  for (int u = 0; u < 4; u++)
    cp16(&Ql[(rr + u * 16) * 64 + c8], qBase + (long)u * 16 * D_MODEL);
  stage_kv(0, 0);

  v16bf aq[2];
  v8f o[4];
  for (int j = 0; j < 4; j++) o[j] = v8f_zero();
  float mx[8], sm[8];
  for (int r = 0; r < 8; r++) { mx[r] = -3.0e38f; sm[r] = 0.0f; }
  const unsigned char* mrow = smask ? smask + (long)b * Lk : nullptr;

  int nt = Lk >> 6;
  for (int t = 0; t < nt; t++) {
    int buf = t & 1;
    int kv0 = t << 6;
    if (t + 1 < nt) {
      stage_kv((t + 1) << 6, buf ^ 1);
      cp_waitN<8>();                  // tile t (and Q) complete; t+1 in flight
    } else {
      cp_waitN<0>();
    }
    __syncthreads();
    if (t == 0)
      for (int ks = 0; ks < 2; ks++)
        aq[ks] = load_a_frag(&Ql[(w * 16 + lm) * 64 + ks * 32], hv);

    // S(16x64) = Q . K^T : B-operand cols are kv rows of Kl (dk-contiguous)
    v8f s[4];
    for (int nb = 0; nb < 4; nb++) {
      s[nb] = v8f_zero();
      for (int ks = 0; ks < 2; ks++) {
        v16bf bK = load_b_frag(&Kl[buf][(nb * 16 + lm) * 64 + ks * 32], hv);
        s[nb] = wmma_bf16(aq[ks], bK, s[nb]);
      }
    }

    // Mask + scale
    float sc[4][8];
    for (int nb = 0; nb < 4; nb++) {
      int kvg = kv0 + nb * 16 + lm;
      for (int r = 0; r < 8; r++) {
        int qg = q0 + w * 16 + r + 8 * hv;
        float sv = s[nb][r] * scale;
        bool ok = causal ? (kvg <= qg) : (mrow ? (mrow[kvg] != 0) : true);
        sc[nb][r] = ok ? sv : -1.0e9f;
      }
    }

    // Online softmax update; P staged to per-wave LDS in [m][kv] order
    for (int r = 0; r < 8; r++) {
      float rowm = sc[0][r];
      for (int nb = 1; nb < 4; nb++) rowm = fmaxf(rowm, sc[nb][r]);
      for (int m = 1; m < 16; m <<= 1) rowm = fmaxf(rowm, __shfl_xor(rowm, m));
      float mnew = fmaxf(mx[r], rowm);
      float corr = __expf(mx[r] - mnew);
      mx[r] = mnew;
      float psum = 0.0f;
      for (int nb = 0; nb < 4; nb++) {
        float p = __expf(sc[nb][r] - mnew);
        psum += p;
        Pl[w][(r + 8 * hv) * 64 + nb * 16 + lm] = (bf16_t)p;
      }
      sm[r] = sm[r] * corr + psum;
      for (int j = 0; j < 4; j++) o[j][r] = o[j][r] * corr;
    }

    // O(16x64) += P(16x64) . V : B-operand cols are dk rows of Vl (kv-contig)
    v16bf ap[2];
    for (int ks = 0; ks < 2; ks++)
      ap[ks] = load_a_frag(&Pl[w][lm * 64 + ks * 32], hv);
    for (int j = 0; j < 4; j++)
      for (int ks = 0; ks < 2; ks++) {
        v16bf bV = load_b_frag(&Vl[buf][(j * 16 + lm) * 64 + ks * 32], hv);
        o[j] = wmma_bf16(ap[ks], bV, o[j]);
      }
    __syncthreads();                  // buf free for tile t+2's fill
  }

  // Normalize by row sum (partial sums reduced across the 16-lane row group)
  for (int r = 0; r < 8; r++) {
    float stot = sm[r];
    for (int m = 1; m < 16; m <<= 1) stot += __shfl_xor(stot, m);
    float inv = 1.0f / stot;
    long rowi = (long)(b * Lq + q0 + w * 16 + r + 8 * hv);
    for (int j = 0; j < 4; j++)
      O[rowi * D_MODEL + h * DK + j * 16 + lm] = (bf16_t)(o[j][r] * inv);
  }
}

// ---------------------------------------------------------------------------
// Orchestration
// ---------------------------------------------------------------------------
extern "C" void kernel_launch(void* const* d_in, const int* in_sizes, int n_in,
                              void* d_out, int out_size, void* d_ws, size_t ws_size,
                              hipStream_t stream) {
  (void)in_sizes; (void)n_in; (void)out_size; (void)ws_size;
  char* ws = (char*)d_ws;
  size_t off = 0;
  auto alloc = [&](size_t bytes) -> void* {
    void* p = ws + off;
    off = (off + bytes + 255) & ~(size_t)255;
    return p;
  };
  auto F = [&](int i) { return (const float*)d_in[i]; };

  // Flattened input index map (params insertion order, then src/tgt/masks)
  const int I_SRC_EMB = 0, I_TGT_EMB = 1, I_GEN_B = 2;
  const int EB = 3, ES = 16;              // enc layer base/stride
  const int DB = 3 + 6 * 16, DS = 26;     // dec layer base/stride  (=99)
  const int I_ENC_NG = DB + 6 * DS;       // 255
  const int I_ENC_NB = I_ENC_NG + 1, I_DEC_NG = I_ENC_NG + 2, I_DEC_NB = I_ENC_NG + 3;
  const int I_SRC = I_ENC_NG + 4, I_TGT = I_SRC + 1, I_SMASK = I_SRC + 2;
  // I_SRC+3 = tgt_mask (causal; computed arithmetically in-kernel)

  // Transposed bf16 weight: W[K][N] fp32 -> WT[N][K] bf16 (K-contiguous)
  auto cvtT = [&](const float* src, int K, int N) -> bf16_t* {
    bf16_t* dst = (bf16_t*)alloc((size_t)K * N * sizeof(bf16_t));
    long n = (long)K * N;
    long blocks = (n + 2047) / 2048;
    if (blocks > 4096) blocks = 4096;
    k_cvt_bf16_T<<<dim3((unsigned)blocks), 256, 0, stream>>>(src, dst, K, N);
    return dst;
  };

  // ---- weights -> bf16 (transposed) ----
  bf16_t *eqw[6], *ekw[6], *evw[6], *eow[6], *ef1[6], *ef2[6];
  for (int l = 0; l < 6; l++) {
    int b = EB + l * ES;
    eqw[l] = cvtT(F(b + 0), 512, 512);
    ekw[l] = cvtT(F(b + 2), 512, 512);
    evw[l] = cvtT(F(b + 4), 512, 512);
    eow[l] = cvtT(F(b + 6), 512, 512);
    ef1[l] = cvtT(F(b + 8), 512, 2048);
    ef2[l] = cvtT(F(b + 10), 2048, 512);
  }
  bf16_t *dsq[6], *dsk[6], *dsv[6], *dso[6], *dcq[6], *dck[6], *dcv[6], *dco[6],
      *df1[6], *df2[6];
  for (int l = 0; l < 6; l++) {
    int b = DB + l * DS;
    dsq[l] = cvtT(F(b + 0), 512, 512);
    dsk[l] = cvtT(F(b + 2), 512, 512);
    dsv[l] = cvtT(F(b + 4), 512, 512);
    dso[l] = cvtT(F(b + 6), 512, 512);
    dcq[l] = cvtT(F(b + 8), 512, 512);
    dck[l] = cvtT(F(b + 10), 512, 512);
    dcv[l] = cvtT(F(b + 12), 512, 512);
    dco[l] = cvtT(F(b + 14), 512, 512);
    df1[l] = cvtT(F(b + 16), 512, 2048);
    df2[l] = cvtT(F(b + 18), 2048, 512);
  }
  // Generator weight = tgt_emb[32000][512], already [N][K]: plain convert.
  bf16_t* embT = (bf16_t*)alloc((size_t)32000 * 512 * sizeof(bf16_t));
  k_cvt_bf16<<<4096, 256, 0, stream>>>(F(I_TGT_EMB), embT, (long)32000 * 512);

  // ---- activations ----
  const int MSRC = BATCH * S_SRC;  // 8192
  const int MTGT = BATCH * S_TGT;  // 4096
  float*  xf   = (float*)alloc((size_t)MSRC * 512 * 4);
  bf16_t* xb   = (bf16_t*)alloc((size_t)MSRC * 512 * 2);
  bf16_t* qb   = (bf16_t*)alloc((size_t)MSRC * 512 * 2);
  bf16_t* kb   = (bf16_t*)alloc((size_t)MSRC * 512 * 2);
  bf16_t* vtb  = (bf16_t*)alloc((size_t)MSRC * 512 * 2);  // [B][H][dk][L]
  bf16_t* ab   = (bf16_t*)alloc((size_t)MSRC * 512 * 2);
  float*  of32 = (float*)alloc((size_t)MSRC * 512 * 4);
  bf16_t* hb   = (bf16_t*)alloc((size_t)MSRC * 2048 * 2);
  bf16_t* memb = (bf16_t*)alloc((size_t)MSRC * 512 * 2);
  bf16_t* kmemb= (bf16_t*)alloc((size_t)MSRC * 512 * 2);
  bf16_t* vmemt= (bf16_t*)alloc((size_t)MSRC * 512 * 2);  // [B][H][dk][S_SRC]
  float*  yf   = (float*)alloc((size_t)MTGT * 512 * 4);
  bf16_t* yb   = (bf16_t*)alloc((size_t)MTGT * 512 * 2);

  auto gemm = [&](const bf16_t* A, const bf16_t* WT, const float* bias, float* Cf,
                  bf16_t* Cb, bf16_t* CbT, int Lseq, int M, int N, int K,
                  int relu) {
    k_gemm<<<dim3(N / 128, M / 128), 256, 0, stream>>>(A, WT, bias, Cf, Cb, CbT,
                                                       Lseq, M, N, K, relu);
  };
  auto addln = [&](const float* x, const float* a, const float* g, const float* bb,
                   float* ofp, bf16_t* ob, int rows) {
    k_add_ln<<<rows, 256, 0, stream>>>(x, a, g, bb, ofp, ob);
  };
  auto flash = [&](const bf16_t* q, const bf16_t* k, const bf16_t* vt, bf16_t* o,
                   int Lq, int Lk, int causal, const unsigned char* sm) {
    k_flash<<<dim3(Lq / 64, N_HEADS, BATCH), 128, 0, stream>>>(q, k, vt, o, Lq, Lk,
                                                               causal, sm, 0.125f);
  };
  const unsigned char* smask = (const unsigned char*)d_in[I_SMASK];

  // ---- encoder ----
  k_embed<<<MSRC, 256, 0, stream>>>((const int*)d_in[I_SRC], F(I_SRC_EMB), xf, xb,
                                    S_SRC);
  for (int l = 0; l < 6; l++) {
    int pb = EB + l * ES;
    gemm(xb, eqw[l], F(pb + 1), nullptr, qb, nullptr, 1, MSRC, 512, 512, 0);
    gemm(xb, ekw[l], F(pb + 3), nullptr, kb, nullptr, 1, MSRC, 512, 512, 0);
    gemm(xb, evw[l], F(pb + 5), nullptr, nullptr, vtb, S_SRC, MSRC, 512, 512, 0);
    flash(qb, kb, vtb, ab, S_SRC, S_SRC, 0, smask);
    gemm(ab, eow[l], F(pb + 7), of32, nullptr, nullptr, 1, MSRC, 512, 512, 0);
    addln(xf, of32, F(pb + 12), F(pb + 13), xf, xb, MSRC);
    gemm(xb, ef1[l], F(pb + 9), nullptr, hb, nullptr, 1, MSRC, 2048, 512, 1);
    gemm(hb, ef2[l], F(pb + 11), of32, nullptr, nullptr, 1, MSRC, 512, 2048, 0);
    addln(xf, of32, F(pb + 14), F(pb + 15), xf, xb, MSRC);
  }
  addln(xf, nullptr, F(I_ENC_NG), F(I_ENC_NB), nullptr, memb, MSRC);

  // ---- decoder ----
  k_embed<<<MTGT, 256, 0, stream>>>((const int*)d_in[I_TGT], F(I_TGT_EMB), yf, yb,
                                    S_TGT);
  for (int l = 0; l < 6; l++) {
    int pb = DB + l * DS;
    gemm(yb, dsq[l], F(pb + 1), nullptr, qb, nullptr, 1, MTGT, 512, 512, 0);
    gemm(yb, dsk[l], F(pb + 3), nullptr, kb, nullptr, 1, MTGT, 512, 512, 0);
    gemm(yb, dsv[l], F(pb + 5), nullptr, nullptr, vtb, S_TGT, MTGT, 512, 512, 0);
    flash(qb, kb, vtb, ab, S_TGT, S_TGT, 1, nullptr);
    gemm(ab, dso[l], F(pb + 7), of32, nullptr, nullptr, 1, MTGT, 512, 512, 0);
    addln(yf, of32, F(pb + 20), F(pb + 21), yf, yb, MTGT);
    gemm(memb, dck[l], F(pb + 11), nullptr, kmemb, nullptr, 1, MSRC, 512, 512, 0);
    gemm(memb, dcv[l], F(pb + 13), nullptr, nullptr, vmemt, S_SRC, MSRC, 512, 512, 0);
    gemm(yb, dcq[l], F(pb + 9), nullptr, qb, nullptr, 1, MTGT, 512, 512, 0);
    flash(qb, kmemb, vmemt, ab, S_TGT, S_SRC, 0, smask);
    gemm(ab, dco[l], F(pb + 15), of32, nullptr, nullptr, 1, MTGT, 512, 512, 0);
    addln(yf, of32, F(pb + 22), F(pb + 23), yf, yb, MTGT);
    gemm(yb, df1[l], F(pb + 17), nullptr, hb, nullptr, 1, MTGT, 2048, 512, 1);
    gemm(hb, df2[l], F(pb + 19), of32, nullptr, nullptr, 1, MTGT, 512, 2048, 0);
    addln(yf, of32, F(pb + 24), F(pb + 25), yf, yb, MTGT);
  }
  addln(yf, nullptr, F(I_DEC_NG), F(I_DEC_NB), nullptr, yb, MTGT);

  // ---- generator: logits = y @ tgt_emb^T + gen_b ----
  gemm(yb, embT, F(I_GEN_B), (float*)d_out, nullptr, nullptr, 1, MTGT, 32000, 512, 0);
}
